// MultiHeadAttention_20005957665365
// MI455X (gfx1250) — matrix operand order repaired and checked by reading, not compiled, not run on hardware
//
#include <hip/hip_runtime.h>
#include <hip/hip_bf16.h>

#define B_  4
#define S_  2048
#define D_  1024
#define H_  16
#define DK_ 64

typedef __bf16 bf16_t;
typedef __attribute__((ext_vector_type(16))) __bf16 v16bf;
typedef __attribute__((ext_vector_type(8)))  __bf16 v8bf;
typedef __attribute__((ext_vector_type(8)))  float  v8f;

#define WMMA_BF16(a, b, c) \
  __builtin_amdgcn_wmma_f32_16x16x32_bf16(false, (a), false, (b), (short)0, (c), false, false)

// ---------------- fragment loaders (layouts per CDNA5 ISA) ----------------

// A 16x32 bf16: lane holds row m=lane&15; elements: k = (e&8)*2 + (lane>>4)*8 + (e&7)
__device__ __forceinline__ v16bf load_frag_a(const bf16_t* p, int ld, int lane) {
  int m = lane & 15, h = lane >> 4;
  const bf16_t* r = p + (long)m * ld + h * 8;
  union { v16bf v; v8bf h2[2]; } u;
  u.h2[0] = *(const v8bf*)(r);
  u.h2[1] = *(const v8bf*)(r + 16);
  return u.v;
}

// Same A layout but source is f32 (convert to bf16 on the fly)
__device__ __forceinline__ v16bf load_frag_a_f32(const float* p, int ld, int lane) {
  int m = lane & 15, h = lane >> 4;
  const float* r = p + (long)m * ld + h * 8;
  v16bf out;
#pragma unroll
  for (int i = 0; i < 8; i++) out[i] = (__bf16)r[i];
#pragma unroll
  for (int i = 0; i < 8; i++) out[8 + i] = (__bf16)r[16 + i];
  return out;
}

// A fragment from LDS f32 scores
__device__ __forceinline__ v16bf load_frag_a_lds(const float* sc, int k0, int lane) {
  int m = lane & 15, h = lane >> 4;
  const float* r = sc + m * S_ + k0 + h * 8;
  v16bf out;
#pragma unroll
  for (int i = 0; i < 8; i++) out[i] = (__bf16)r[i];
#pragma unroll
  for (int i = 0; i < 8; i++) out[8 + i] = (__bf16)r[16 + i];
  return out;
}

// B 32x16 bf16 from an [n][k] row-major buffer: lane holds col n=lane&15;
// elements: k = (e&8)*2 + (lane>>4)*8 + (e&7) (same order as A)
__device__ __forceinline__ v16bf load_frag_b(const bf16_t* p, int ld, int lane) {
  int n = lane & 15, h = lane >> 4;
  const bf16_t* r = p + (long)n * ld + h * 8;
  union { v16bf v; v8bf h2[2]; } u;
  u.h2[0] = *(const v8bf*)(r);
  u.h2[1] = *(const v8bf*)(r + 16);
  return u.v;
}

// ---------------- kernels ----------------

__global__ void cvt_f32_bf16(const float* __restrict__ src, bf16_t* __restrict__ dst, int n) {
  int i = blockIdx.x * blockDim.x + threadIdx.x;
  if (i < n) dst[i] = (bf16_t)src[i];
}

// Y(bf16) = bf16( X(f32)[M,K] @ Wb(bf16)[N,K]^T + bias ), M = B*S, N = K = D
// vtrans: write v transposed per head into vT[B][H][DK][S]
__global__ void proj_kernel(const float* __restrict__ X, const bf16_t* __restrict__ Wb,
                            const float* __restrict__ bias, bf16_t* __restrict__ Y,
                            int vtrans) {
  const int lane = threadIdx.x & 31;
  const int wave = threadIdx.x >> 5;       // 0..7
  const int m0 = blockIdx.y * 32 + (wave >> 2) * 16;
  const int n0 = blockIdx.x * 64 + (wave & 3) * 16;
  const float*  Arow = X  + (long)m0 * D_ + (long)(lane & 15) * D_;
  const bf16_t* Brow = Wb + (long)n0 * D_ + (long)(lane & 15) * D_;
  v8f acc = {};
  for (int k = 0; k < D_; k += 32) {
    // prefetch next k-chunk (global_prefetch_b8; no counter pressure)
    if (k + 32 < D_) {
      __builtin_prefetch(Arow + k + 32, 0, 0);
      __builtin_prefetch(Brow + k + 32, 0, 0);
    }
    v16bf a = load_frag_a_f32(X + (long)m0 * D_ + k, D_, lane);
    v16bf b = load_frag_b(Wb + (long)n0 * D_ + k, D_, lane);
    acc = WMMA_BF16(a, b, acc);
  }
  const int n = n0 + (lane & 15);
  const int half = lane >> 4;
  const float bv = bias[n];
  if (!vtrans) {
#pragma unroll
    for (int v = 0; v < 8; v++) {
      int m = m0 + half * 8 + v;
      Y[(long)m * D_ + n] = (bf16_t)(acc[v] + bv);
    }
  } else {
    int h  = n >> 6;   // n / DK
    int dk = n & 63;
#pragma unroll
    for (int v = 0; v < 8; v++) {
      int m  = m0 + half * 8 + v;
      int bb = m >> 11;          // m / S
      int s  = m & (S_ - 1);
      Y[(((long)bb * H_ + h) * DK_ + dk) * S_ + s] = (bf16_t)(acc[v] + bv);
    }
  }
}

// out(f32) = Xb(bf16)[M,K] @ Wob(bf16)[N,K]^T + bias
__global__ void outproj_kernel(const bf16_t* __restrict__ Xb, const bf16_t* __restrict__ Wb,
                               const float* __restrict__ bias, float* __restrict__ Y) {
  const int lane = threadIdx.x & 31;
  const int wave = threadIdx.x >> 5;
  const int m0 = blockIdx.y * 32 + (wave >> 2) * 16;
  const int n0 = blockIdx.x * 64 + (wave & 3) * 16;
  const bf16_t* Arow = Xb + (long)m0 * D_ + (long)(lane & 15) * D_;
  const bf16_t* Brow = Wb + (long)n0 * D_ + (long)(lane & 15) * D_;
  v8f acc = {};
  for (int k = 0; k < D_; k += 32) {
    if (k + 32 < D_) {
      __builtin_prefetch(Arow + k + 32, 0, 0);
      __builtin_prefetch(Brow + k + 32, 0, 0);
    }
    v16bf a = load_frag_a(Xb + (long)m0 * D_ + k, D_, lane);
    v16bf b = load_frag_b(Wb + (long)n0 * D_ + k, D_, lane);
    acc = WMMA_BF16(a, b, acc);
  }
  const int n = n0 + (lane & 15);
  const int half = lane >> 4;
  const float bv = bias[n];
#pragma unroll
  for (int v = 0; v < 8; v++) {
    int m = m0 + half * 8 + v;
    Y[(long)m * D_ + n] = acc[v] + bv;
  }
}

// One workgroup (4 waves) per (b, h, 16-row q tile).
// Dynamic LDS: float sc[16][S] = 128 KB (CDNA5 WGP has 320 KB -> 2 WGs resident).
#define ATT_THREADS 128
__global__ void attn_kernel(const bf16_t* __restrict__ qb, const bf16_t* __restrict__ kb,
                            const bf16_t* __restrict__ vtb,
                            float* __restrict__ attn_out, bf16_t* __restrict__ xb) {
  extern __shared__ float sc[];     // [16][S_]
  const int qt = blockIdx.x;        // 0..S/16-1
  const int h  = blockIdx.y;
  const int b  = blockIdx.z;
  const int q0 = qt * 16;
  const int lane = threadIdx.x & 31;
  const int wave = threadIdx.x >> 5;     // 0..3
  const int nkt = qt + 1;                // causal: key tiles 0..qt
  const int nk  = nkt * 16;
  const int nkpad = (nk + 31) & ~31;

  const bf16_t* qbase = qb + ((long)(b * S_ + q0)) * D_ + h * DK_;
  const bf16_t* kbase = kb + ((long)b * S_) * D_ + h * DK_;

  // q tile fragments for the two K=32 chunks of DK=64
  v16bf qa0 = load_frag_a(qbase + 0,  D_, lane);
  v16bf qa1 = load_frag_a(qbase + 32, D_, lane);

  // scores = q @ k^T * 1/sqrt(DK), causal mask, into LDS
  for (int kt = wave; kt < nkt; kt += 4) {
    const bf16_t* kp = kbase + (long)(kt * 16) * D_;
    if (kt + 4 < nkt)
      __builtin_prefetch(kbase + (long)((kt + 4) * 16 + (lane & 15)) * D_, 0, 0);
    v8f acc = {};
    v16bf b0 = load_frag_b(kp + 0,  D_, lane);
    v16bf b1 = load_frag_b(kp + 32, D_, lane);
    acc = WMMA_BF16(qa0, b0, acc);
    acc = WMMA_BF16(qa1, b1, acc);
    int n = lane & 15, half = lane >> 4;
    int kg = kt * 16 + n;
#pragma unroll
    for (int v = 0; v < 8; v++) {
      int m  = half * 8 + v;
      int qg = q0 + m;
      float s = acc[v] * 0.125f;       // 1/sqrt(64)
      if (kg > qg) s = -1e30f;
      sc[m * S_ + kt * 16 + n] = s;
    }
  }
  __syncthreads();

  // softmax over [0, nk): 8 threads per row, shuffle reduce
  {
    int row = threadIdx.x >> 3;       // 0..15
    int sub = threadIdx.x & 7;
    float mx = -1e30f;
    for (int c = sub; c < nk; c += 8) mx = fmaxf(mx, sc[row * S_ + c]);
#pragma unroll
    for (int off = 1; off < 8; off <<= 1) mx = fmaxf(mx, __shfl_xor(mx, off, 32));
    float sum = 0.f;
    for (int c = sub; c < nk; c += 8) {
      float p = __expf(sc[row * S_ + c] - mx);
      sc[row * S_ + c] = p;
      sum += p;
    }
#pragma unroll
    for (int off = 1; off < 8; off <<= 1) sum += __shfl_xor(sum, off, 32);
    float inv = 1.f / sum;
    for (int c = sub; c < nk; c += 8) sc[row * S_ + c] *= inv;
  }
  __syncthreads();

  // coalesced f32 attn store (zeros in masked region), zero LDS pad tail
  {
    float* arow = attn_out + ((long)(b * H_ + h) * S_ + q0) * S_;
    for (int idx = threadIdx.x; idx < 16 * S_; idx += ATT_THREADS) {
      int col = idx & (S_ - 1);
      float val;
      if (col < nk) val = sc[idx];
      else { val = 0.f; if (col < nkpad) sc[idx] = 0.f; }
      arow[((long)(idx >> 11)) * S_ + col] = val;
    }
  }
  __syncthreads();

  // x = attn @ v : wave w computes dk tile n0 = 16*w (DK=64 total)
  {
    int n0 = wave * 16;
    const bf16_t* vbase = vtb + (((long)(b * H_ + h)) * DK_ + n0) * S_;
    v8f acc = {};
    for (int s0 = 0; s0 < nkpad; s0 += 32) {
      v16bf a  = load_frag_a_lds(sc, s0, lane);
      v16bf bb = load_frag_b(vbase + s0, S_, lane);
      acc = WMMA_BF16(a, bb, acc);
    }
    int n = lane & 15, half = lane >> 4;
#pragma unroll
    for (int v = 0; v < 8; v++) {
      int m = q0 + half * 8 + v;
      xb[((long)(b * S_ + m)) * D_ + h * DK_ + n0 + n] = (bf16_t)acc[v];
    }
  }
}

// ---------------- launcher ----------------

extern "C" void kernel_launch(void* const* d_in, const int* in_sizes, int n_in,
                              void* d_out, int out_size, void* d_ws, size_t ws_size,
                              hipStream_t stream) {
  (void)in_sizes; (void)n_in; (void)out_size; (void)ws_size;
  const float* Q  = (const float*)d_in[0];
  const float* K  = (const float*)d_in[1];
  const float* V  = (const float*)d_in[2];
  // d_in[3] = mask: causal tril, implemented analytically
  const float* Wq = (const float*)d_in[4];
  const float* bq = (const float*)d_in[5];
  const float* Wk = (const float*)d_in[6];
  const float* bk = (const float*)d_in[7];
  const float* Wv = (const float*)d_in[8];
  const float* bv = (const float*)d_in[9];
  const float* Wo = (const float*)d_in[10];
  const float* bo = (const float*)d_in[11];

  const size_t DD   = (size_t)D_ * D_;            // 1M elems
  const size_t BSD  = (size_t)B_ * S_ * D_;       // 8M elems
  char* ws = (char*)d_ws;
  bf16_t* wqb = (bf16_t*)(ws + 0 * DD * 2);
  bf16_t* wkb = (bf16_t*)(ws + 1 * DD * 2);
  bf16_t* wvb = (bf16_t*)(ws + 2 * DD * 2);
  bf16_t* wob = (bf16_t*)(ws + 3 * DD * 2);
  char* big = ws + 4 * DD * 2;
  bf16_t* qbuf = (bf16_t*)(big + 0 * BSD * 2);
  bf16_t* kbuf = (bf16_t*)(big + 1 * BSD * 2);
  bf16_t* vtb  = (bf16_t*)(big + 2 * BSD * 2);    // [B][H][DK][S]
  bf16_t* xb   = (bf16_t*)(big + 3 * BSD * 2);

  float* out  = (float*)d_out;
  float* attn = out + BSD;

  // 1) weights f32 -> bf16
  int nW = (int)DD, cvtBlocks = (nW + 255) / 256;
  cvt_f32_bf16<<<cvtBlocks, 256, 0, stream>>>(Wq, wqb, nW);
  cvt_f32_bf16<<<cvtBlocks, 256, 0, stream>>>(Wk, wkb, nW);
  cvt_f32_bf16<<<cvtBlocks, 256, 0, stream>>>(Wv, wvb, nW);
  cvt_f32_bf16<<<cvtBlocks, 256, 0, stream>>>(Wo, wob, nW);

  // 2) projections (M=B*S=8192, N=K=D=1024); block = 8 waves -> 32x64 tile
  dim3 pg(D_ / 64, (B_ * S_) / 32);
  proj_kernel<<<pg, 256, 0, stream>>>(Q, wqb, bq, qbuf, 0);
  proj_kernel<<<pg, 256, 0, stream>>>(K, wkb, bk, kbuf, 0);
  proj_kernel<<<pg, 256, 0, stream>>>(V, wvb, bv, vtb, 1);

  // 3) attention: one block per (q-tile, head, batch); 128 KB dynamic LDS
  attn_kernel<<<dim3(S_ / 16, H_, B_), ATT_THREADS, 16 * S_ * sizeof(float), stream>>>(
      qbuf, kbuf, vtb, attn, xb);

  // 4) output projection -> f32 out
  outproj_kernel<<<pg, 256, 0, stream>>>(xb, wob, bo, out);
}
